// Weird_Attention_80934363725856
// MI455X (gfx1250) — matrix operand order repaired and checked by reading, not compiled, not run on hardware
//
#include <hip/hip_runtime.h>

// ---------------------------------------------------------------------------
// CDNA5 (gfx1250) WMMA implementation of the "weird attention" block.
//
// Algebraic restructuring (matmul associativity):
//   glb_ctx + plb_ctx = glb_probs @ (gv + softmax(pq pk^T) @ pv)
// -> two standard flash-style attentions + 7 GEMMs; no O(S^3) bridge matmul,
//    no materialized S x S probability matrices.
//
// Precision: bf16 operands, fp32 accumulation via v_wmma_f32_16x16x32_bf16.
// ---------------------------------------------------------------------------

#define D_MODEL 1024
#define SEQ     1024
#define BATCH   4
#define M_ROWS  (BATCH * SEQ)   // 4096

typedef __attribute__((ext_vector_type(16))) __bf16        v16bf;
typedef __attribute__((ext_vector_type(8)))  float         v8f;
typedef __attribute__((ext_vector_type(4)))  unsigned int  v4u;

union Frag { v4u u[2]; v16bf b; };

__device__ __forceinline__ unsigned short f2bf(float f) {
  unsigned int u = __float_as_uint(f);
  u = (u + 0x7FFFu + ((u >> 16) & 1u)) >> 16;   // round-to-nearest-even
  return (unsigned short)u;
}
__device__ __forceinline__ float bf2f(unsigned short b) {
  return __uint_as_float(((unsigned int)b) << 16);
}

// ---------------- elementwise: f32 -> bf16 ----------------
__global__ __launch_bounds__(256) void cvt_f32_bf16(const float* __restrict__ src,
                                                    unsigned short* __restrict__ dst,
                                                    int n) {
  for (int i = blockIdx.x * blockDim.x + threadIdx.x; i < n; i += gridDim.x * blockDim.x)
    dst[i] = f2bf(src[i]);
}

// ---------------- elementwise: vmod = a + b (bf16) ----------------
__global__ __launch_bounds__(256) void add_bf16(const unsigned short* __restrict__ a,
                                                const unsigned short* __restrict__ b,
                                                unsigned short* __restrict__ o,
                                                int n) {
  for (int i = blockIdx.x * blockDim.x + threadIdx.x; i < n; i += gridDim.x * blockDim.x)
    o[i] = f2bf(bf2f(a[i]) + bf2f(b[i]));
}

// ---------------------------------------------------------------------------
// C[M x 1024] = A[M x 1024] @ W[1024 x 1024]^T + bias   (torch Linear)
// Both A rows and W rows are contiguous along K -> direct 16B fragment loads.
// Block: 256 threads = 8 waves; wave tile 32x64; block tile 128x128.
// ---------------------------------------------------------------------------
template <bool F32OUT>
__global__ __launch_bounds__(256) void gemm_bias_wmma(
    const unsigned short* __restrict__ A,     // bf16 [M x 1024]
    const unsigned short* __restrict__ W,     // bf16 [1024(out) x 1024(in)]
    const float* __restrict__ bias,           // f32  [1024]
    void* __restrict__ Cout) {
  const int lane = threadIdx.x & 31;
  const int wave = threadIdx.x >> 5;
  const int wm = wave & 3;                    // 4 waves along M
  const int wn = wave >> 2;                   // 2 waves along N
  const int mBase = blockIdx.y * 128 + wm * 32;
  const int nBase = blockIdx.x * 128 + wn * 64;

  const int l15  = lane & 15;
  const int hiA  = (lane >> 4) * 8;           // A-fragment K offset (interleaved halves)
  const int hiB  = (lane >> 4) * 8;           // B-fragment K offset (interleaved halves)

  const v8f vzero = {0.f, 0.f, 0.f, 0.f, 0.f, 0.f, 0.f, 0.f};
  v8f acc[2][4];
#pragma unroll
  for (int mi = 0; mi < 2; ++mi)
#pragma unroll
    for (int ni = 0; ni < 4; ++ni) acc[mi][ni] = vzero;

  for (int k0 = 0; k0 < 1024; k0 += 32) {
    Frag af[2];
#pragma unroll
    for (int mi = 0; mi < 2; ++mi) {
      const unsigned short* p = A + (mBase + mi * 16 + l15) * 1024 + k0 + hiA;
      af[mi].u[0] = *(const v4u*)p;           // K = k0+hiA .. +7
      af[mi].u[1] = *(const v4u*)(p + 16);    // K = k0+16+hiA .. +7
    }
    Frag bfr[4];
#pragma unroll
    for (int ni = 0; ni < 4; ++ni) {
      const unsigned short* p = W + (nBase + ni * 16 + l15) * 1024 + k0 + hiB;
      bfr[ni].u[0] = *(const v4u*)p;          // K = k0+hiB .. +7
      bfr[ni].u[1] = *(const v4u*)(p + 16);   // K = k0+16+hiB .. +7
    }
#pragma unroll
    for (int mi = 0; mi < 2; ++mi)
#pragma unroll
      for (int ni = 0; ni < 4; ++ni)
        acc[mi][ni] = __builtin_amdgcn_wmma_f32_16x16x32_bf16(
            false, af[mi].b, false, bfr[ni].b, (short)0, acc[mi][ni], false, false);
  }

  const int rowHi = (lane >> 4) * 8;          // C/D layout: VGPR r -> M = r + 8*(lane>>4)
#pragma unroll
  for (int mi = 0; mi < 2; ++mi) {
#pragma unroll
    for (int ni = 0; ni < 4; ++ni) {
      const int col = nBase + ni * 16 + l15;
      const float bz = bias[col];
#pragma unroll
      for (int r = 0; r < 8; ++r) {
        const int row = mBase + mi * 16 + rowHi + r;
        const float v = acc[mi][ni][r] + bz;
        if (F32OUT) ((float*)Cout)[row * 1024 + col] = v;
        else        ((unsigned short*)Cout)[row * 1024 + col] = f2bf(v);
      }
    }
  }
}

// ---------------------------------------------------------------------------
// Flash-style attention per head:  O = softmax(Q K^T * scale) V
// Q/K/V/O live in merged [B, S, 1024] bf16 buffers; head h occupies cols
// h*64 .. h*64+63.  One block = (b,h) x 128 queries; wave owns 16 queries.
// K block + transposed V block staged in LDS; P tile through per-wave LDS.
// ---------------------------------------------------------------------------
__global__ __launch_bounds__(256) void attn_wmma(
    const unsigned short* __restrict__ Q,
    const unsigned short* __restrict__ Kp,
    const unsigned short* __restrict__ V,
    unsigned short* __restrict__ O,
    float scale) {
  __shared__ unsigned int   KldsU[64 * 36];     // K block [key][d], stride 72 halves
  __shared__ unsigned short Vt[64 * 72];        // V block transposed [d][key]
  __shared__ unsigned short Plds[8 * 16 * 72];  // per-wave P tiles [16 q][64 key]

  const int lane = threadIdx.x & 31;
  const int wave = threadIdx.x >> 5;
  const int l15  = lane & 15;
  const int hi8  = (lane >> 4) * 8;
  const int hi16 = (lane >> 4) * 16;

  const int bh   = blockIdx.y;                  // b*16 + h
  const long base = (long)(bh >> 4) * SEQ * 1024 + (bh & 15) * 64;

  const int q0 = blockIdx.x * 128 + wave * 16;

  // Q fragments pinned in registers (d-chunks 0..31, 32..63)
  Frag qf[2];
  {
    const unsigned short* qr = Q + base + (long)(q0 + l15) * 1024;
#pragma unroll
    for (int c = 0; c < 2; ++c) {
      const unsigned short* p = qr + c * 32 + hi8;
      qf[c].u[0] = *(const v4u*)p;
      qf[c].u[1] = *(const v4u*)(p + 16);
    }
  }

  const v8f vzero = {0.f, 0.f, 0.f, 0.f, 0.f, 0.f, 0.f, 0.f};
  float m[8], l[8];
#pragma unroll
  for (int r = 0; r < 8; ++r) { m[r] = -3.0e38f; l[r] = 0.0f; }
  v8f o[4];
#pragma unroll
  for (int n = 0; n < 4; ++n) o[n] = vzero;

  const unsigned short* KldsH = (const unsigned short*)KldsU;
  const int pBase = wave * 16 * 72;

  for (int kb = 0; kb < SEQ; kb += 64) {
    // prefetch hint for next K block (wraps to stay in-bounds)
    __builtin_prefetch(Kp + base + (long)((kb + 64) & (SEQ - 1)) * 1024, 0, 1);

    __syncthreads();                            // protect LDS reuse across iterations
    // cooperative stage: K block row-major, V block transposed (2048 u32 pairs)
#pragma unroll
    for (int it = 0; it < 8; ++it) {
      const int idx = threadIdx.x + it * 256;
      const int r   = idx >> 5;                 // key row 0..63
      const int cp  = idx & 31;                 // d-pair 0..31
      const long grow = base + (long)(kb + r) * 1024;
      KldsU[r * 36 + cp] = ((const unsigned int*)(Kp + grow))[cp];
      const unsigned int vv = ((const unsigned int*)(V + grow))[cp];
      const int d0 = cp * 2;
      Vt[d0 * 72 + r]       = (unsigned short)(vv & 0xFFFFu);
      Vt[(d0 + 1) * 72 + r] = (unsigned short)(vv >> 16);
    }
    __syncthreads();

    // scores: 16 queries x 64 keys as four 16x16 WMMA tiles (2 K-steps each)
    v8f st[4];
#pragma unroll
    for (int t = 0; t < 4; ++t) {
      st[t] = vzero;
#pragma unroll
      for (int c = 0; c < 2; ++c) {
        Frag kf;
        const unsigned short* p = KldsH + (t * 16 + l15) * 72 + c * 32 + hi8;
        kf.u[0] = *(const v4u*)p;
        kf.u[1] = *(const v4u*)(p + 16);
        st[t] = __builtin_amdgcn_wmma_f32_16x16x32_bf16(
            false, qf[c].b, false, kf.b, (short)0, st[t], false, false);
      }
      st[t] = st[t] * scale;
    }

    // online softmax over the 16x64 strip (row stats replicated per half-wave)
    float fac[8], rs[8];
#pragma unroll
    for (int r = 0; r < 8; ++r) {
      float x = st[0][r];
#pragma unroll
      for (int t = 1; t < 4; ++t) x = fmaxf(x, st[t][r]);
#pragma unroll
      for (int msk = 1; msk < 16; msk <<= 1) x = fmaxf(x, __shfl_xor(x, msk, 32));
      const float mn = fmaxf(m[r], x);
      fac[r] = __expf(m[r] - mn);
      m[r] = mn;
      rs[r] = 0.f;
    }
#pragma unroll
    for (int t = 0; t < 4; ++t)
#pragma unroll
      for (int r = 0; r < 8; ++r) {
        const float p = __expf(st[t][r] - m[r]);
        st[t][r] = p;
        rs[r] += p;
      }
#pragma unroll
    for (int r = 0; r < 8; ++r) {
      float x = rs[r];
#pragma unroll
      for (int msk = 1; msk < 16; msk <<= 1) x += __shfl_xor(x, msk, 32);
      l[r] = l[r] * fac[r] + x;
#pragma unroll
      for (int n = 0; n < 4; ++n) o[n][r] *= fac[r];
    }

    // P (bf16) -> per-wave LDS tile, then reload as WMMA A-fragments
#pragma unroll
    for (int t = 0; t < 4; ++t)
#pragma unroll
      for (int r = 0; r < 8; ++r)
        Plds[pBase + (hi8 + r) * 72 + t * 16 + l15] = f2bf(st[t][r]);
    // wave-local RAW through LDS: DS ops from one wave complete in order,
    // but wait for dscnt before the cross-lane fragment reads.
    asm volatile("s_wait_dscnt 0x0" ::: "memory");

    Frag pf[2];
#pragma unroll
    for (int c = 0; c < 2; ++c) {
      const unsigned short* p = Plds + pBase + l15 * 72 + c * 32 + hi8;
      pf[c].u[0] = *(const v4u*)p;
      pf[c].u[1] = *(const v4u*)(p + 16);
    }
    // O += P @ V  (V read transposed: contiguous along key)
#pragma unroll
    for (int n = 0; n < 4; ++n)
#pragma unroll
      for (int c = 0; c < 2; ++c) {
        Frag vf;
        const unsigned short* p = Vt + (n * 16 + l15) * 72 + c * 32 + hi8;
        vf.u[0] = *(const v4u*)p;
        vf.u[1] = *(const v4u*)(p + 16);
        o[n] = __builtin_amdgcn_wmma_f32_16x16x32_bf16(
            false, pf[c].b, false, vf.b, (short)0, o[n], false, false);
      }
  }

  // normalize and store bf16 into merged [B,S,1024] layout
#pragma unroll
  for (int n = 0; n < 4; ++n)
#pragma unroll
    for (int r = 0; r < 8; ++r) {
      const int row = q0 + hi8 + r;
      const float v = o[n][r] / l[r];
      O[base + (long)row * 1024 + n * 16 + l15] = f2bf(v);
    }
}

// ---------------------------------------------------------------------------
extern "C" void kernel_launch(void* const* d_in, const int* in_sizes, int n_in,
                              void* d_out, int out_size, void* d_ws, size_t ws_size,
                              hipStream_t stream) {
  (void)in_sizes; (void)n_in; (void)out_size; (void)ws_size;

  const float* feat[3] = {(const float*)d_in[0], (const float*)d_in[1], (const float*)d_in[2]};
  const float* Wf[7];
  const float* Bf[7];
  for (int i = 0; i < 7; ++i) {
    Wf[i] = (const float*)d_in[3 + 2 * i];   // glb_q, glb_k, glb_v, plb_q, plb_k, plb_v, dense
    Bf[i] = (const float*)d_in[4 + 2 * i];
  }

  // workspace carve-up (bf16 elements)
  unsigned short* ws = (unsigned short*)d_ws;
  const size_t FEAT = (size_t)M_ROWS * D_MODEL;   // 4 Mi
  const size_t WSZ  = (size_t)D_MODEL * D_MODEL;  // 1 Mi
  unsigned short* featB[3] = {ws, ws + FEAT, ws + 2 * FEAT};
  unsigned short* Wb[7];
  for (int i = 0; i < 7; ++i) Wb[i] = ws + 3 * FEAT + i * WSZ;
  unsigned short* proj[6];
  for (int j = 0; j < 6; ++j) proj[j] = ws + 3 * FEAT + 7 * WSZ + j * FEAT;
  unsigned short* inner = ws + 3 * FEAT + 7 * WSZ + 6 * FEAT;
  unsigned short* vmod  = inner + FEAT;
  unsigned short* ctx   = vmod + FEAT;

  // 1) downconvert activations and weights to bf16
  for (int i = 0; i < 3; ++i)
    cvt_f32_bf16<<<512, 256, 0, stream>>>(feat[i], featB[i], (int)FEAT);
  for (int i = 0; i < 7; ++i)
    cvt_f32_bf16<<<256, 256, 0, stream>>>(Wf[i], Wb[i], (int)WSZ);

  // 2) six QKV projections: proj = feat @ W^T + b
  const dim3 gGrid(D_MODEL / 128, M_ROWS / 128);       // (8, 32)
  const int src[6] = {0, 1, 1, 1, 2, 2};               // gq<-G; gk,gv,pq<-L; pk,pv<-T
  for (int j = 0; j < 6; ++j)
    gemm_bias_wmma<false><<<gGrid, 256, 0, stream>>>(featB[src[j]], Wb[j], Bf[j], proj[j]);

  const float scale = 0.125f;                          // 1/sqrt(64)
  const dim3 aGrid(SEQ / 128, BATCH * 16);             // (8, 64)

  // 3) inner = softmax(pq pk^T) @ pv
  attn_wmma<<<aGrid, 256, 0, stream>>>(proj[3], proj[4], proj[5], inner, scale);
  // 4) vmod = gv + inner   (folds the bridge matmul via associativity)
  add_bf16<<<1024, 256, 0, stream>>>(proj[2], inner, vmod, (int)FEAT);
  // 5) ctx = softmax(gq gk^T) @ vmod  == merge(glb_ctx) + merge(plb_ctx)
  attn_wmma<<<aGrid, 256, 0, stream>>>(proj[0], proj[1], vmod, ctx, scale);
  // 6) out = ctx @ dense_w^T + dense_b   (fp32 output)
  gemm_bias_wmma<true><<<gGrid, 256, 0, stream>>>(ctx, Wb[6], Bf[6], d_out);
}
